// AttentionLayer_22368189677634
// MI455X (gfx1250) — matrix operand order repaired and checked by reading, not compiled, not run on hardware
//
#include <hip/hip_runtime.h>
#include <hip/hip_bf16.h>
#include <cstdint>
#include <cstddef>

// ---------------- types / WMMA helpers (gfx1250) ----------------
typedef __bf16 bf16_t;
typedef __attribute__((ext_vector_type(8)))  __bf16 v8bf;
typedef __attribute__((ext_vector_type(16))) __bf16 v16bf;
typedef __attribute__((ext_vector_type(8)))  float  v8f;

union AFrag { v16bf v; v8bf h[2]; };

__device__ __forceinline__ v8f wmma_bf16(const AFrag& a, const AFrag& b, v8f c) {
    // D(16x16 f32) = A(16x32 bf16) x B(32x16 bf16) + C
    return __builtin_amdgcn_wmma_f32_16x16x32_bf16(
        /*neg_a=*/false, a.v, /*neg_b=*/false, b.v,
        /*c_mod=*/(short)0, c, /*reuse_a=*/false, /*reuse_b=*/false);
}

// ---------------- problem constants ----------------
#define NTOK  1576          // T*QHW
#define DIM   768
#define DIM3  2304
#define NHD   12
#define HD    64
#define QHW   197
#define TT    8
#define HW    196
#define SPAD  1600          // padded key dim (50 * 32)
#define MROWS 3152          // B*NTOK
#define SCALE 0.125f        // 64^-0.5

// element counts
#define E_X   (MROWS*DIM)          // 2,420,736  (query / key_value / Q / K / Xattn)
#define E_WQ  (DIM3*DIM)           // 1,769,472
#define E_WP  (DIM*DIM)            //   589,824
#define E_VT  (2*NHD*HD*SPAD)      // 2,457,600
#define E_RH  (2*NHD*TT*HW*14)     //   526,848
#define E_RT  (2*NHD*TT*QHW*TT)    //   302,592

// ============================================================
// 1) convert f32 -> bf16 for activations/weights; zero-fill Vt
// ============================================================
__global__ void convert_kernel(const float* __restrict__ query,
                               const float* __restrict__ key_value,
                               const float* __restrict__ wqkv,
                               const float* __restrict__ wproj,
                               bf16_t* __restrict__ Xq, bf16_t* __restrict__ Xkv,
                               bf16_t* __restrict__ Wq, bf16_t* __restrict__ Wp,
                               bf16_t* __restrict__ Vt)
{
    const long total = (long)E_X + E_X + E_WQ + E_WP + E_VT;
    for (long i = (long)blockIdx.x * blockDim.x + threadIdx.x; i < total;
         i += (long)gridDim.x * blockDim.x) {
        long j = i;
        if (j < E_X)            { Xq[j]  = (bf16_t)query[j];     continue; }
        j -= E_X;
        if (j < E_X)            { Xkv[j] = (bf16_t)key_value[j]; continue; }
        j -= E_X;
        if (j < E_WQ)           { Wq[j]  = (bf16_t)wqkv[j];      continue; }
        j -= E_WQ;
        if (j < E_WP)           { Wp[j]  = (bf16_t)wproj[j];     continue; }
        j -= E_WP;
        Vt[j] = (bf16_t)0.0f;   // zero pad region + everything (rewritten later)
    }
}

// ============================================================
// 2) QKV projection GEMM (NT, bf16 WMMA, f32 accum)
//    out[R, O] = sum_c A[R,c] * Wqkv[O,c] + b_qkv[O]
//    R in [0,3152), O in [0,2304); A = Xq for O<768 else Xkv
//    Scatter: q -> Qbf + Qf32 [b,h,n,c]; k -> Kbf [b,h,n,c];
//             v -> Vt [b,h,c,n(SPAD)]
// ============================================================
__global__ __launch_bounds__(256) void qkv_gemm_kernel(
    const bf16_t* __restrict__ Xq, const bf16_t* __restrict__ Xkv,
    const bf16_t* __restrict__ W,  const float* __restrict__ bqkv,
    bf16_t* __restrict__ Qbf, float* __restrict__ Qf,
    bf16_t* __restrict__ Kbf, bf16_t* __restrict__ Vt)
{
    const int MT = MROWS / 16;        // 197
    const int NG = DIM3 / 64;         // 36
    int gwave = blockIdx.x * 8 + (threadIdx.x >> 5);
    if (gwave >= MT * NG) return;
    int mt = gwave % MT, ng = gwave / MT;
    int lane = threadIdx.x & 31, ln = lane & 15, half = lane >> 4;
    int n0 = ng * 64;

    const bf16_t* A = (n0 < DIM) ? Xq : Xkv;
    const bf16_t* arow = A + (size_t)(mt * 16 + ln) * DIM;

    v8f acc[4] = {};
    for (int k0 = 0; k0 < DIM; k0 += 32) {
        AFrag a;
        a.h[0] = *(const v8bf*)(arow + k0 + half * 8);
        a.h[1] = *(const v8bf*)(arow + k0 + 16 + half * 8);
        #pragma unroll
        for (int g = 0; g < 4; ++g) {
            AFrag bfr;
            const bf16_t* brow = W + (size_t)(n0 + g * 16 + ln) * DIM + k0;
            bfr.h[0] = *(const v8bf*)(brow + half * 8);
            bfr.h[1] = *(const v8bf*)(brow + 16 + half * 8);
            acc[g] = wmma_bf16(a, bfr, acc[g]);
        }
    }
    #pragma unroll
    for (int g = 0; g < 4; ++g) {
        int O = n0 + g * 16 + ln;
        int which = O / DIM;
        int rem = O % DIM;
        int h = rem >> 6, c = rem & 63;
        float bias = bqkv[O];
        #pragma unroll
        for (int r = 0; r < 8; ++r) {
            int R = mt * 16 + half * 8 + r;
            int bb = R / NTOK, n = R % NTOK;
            float val = acc[g][r] + bias;
            size_t idx = ((size_t)(bb * NHD + h) * NTOK + n) * HD + c;
            if (which == 0)      { Qbf[idx] = (bf16_t)val; Qf[idx] = val; }
            else if (which == 1) { Kbf[idx] = (bf16_t)val; }
            else { Vt[((size_t)(bb * NHD + h) * HD + c) * SPAD + n] = (bf16_t)val; }
        }
    }
}

// ============================================================
// 3) decomposed rel-pos bias precompute (small VALU kernel)
//    relH[bh,t,hw,kh] = q(t,hw) . rel_pos_h[qh-kh+13]
//    relW[bh,t,hw,kw] = q(t,hw) . rel_pos_w[qw-kw+13]
//    relT[bh,t,qi,kt] = q(t,qi) . rel_pos_t[t-kt+7]
// ============================================================
__global__ void relbias_kernel(const float* __restrict__ Qf,
                               const float* __restrict__ rph,
                               const float* __restrict__ rpw,
                               const float* __restrict__ rpt,
                               float* __restrict__ relH,
                               float* __restrict__ relW,
                               float* __restrict__ relT)
{
    long i = (long)blockIdx.x * blockDim.x + threadIdx.x;
    const long total = 2L * E_RH + E_RT;
    if (i >= total) return;
    if (i < 2L * E_RH) {
        bool isH = (i < E_RH);
        long j = isH ? i : i - E_RH;
        int kx = (int)(j % 14); long t0 = j / 14;
        int hw = (int)(t0 % HW); t0 /= HW;
        int t  = (int)(t0 % TT); int bh = (int)(t0 / TT);
        int qx = isH ? (hw / 14) : (hw % 14);
        int tok = t * QHW + 1 + hw;
        const float* q = Qf + ((size_t)bh * NTOK + tok) * HD;
        const float* r = (isH ? rph : rpw) + (size_t)(qx - kx + 13) * HD;
        float s = 0.f;
        for (int c = 0; c < HD; ++c) s += q[c] * r[c];
        if (isH) relH[j] = s; else relW[j] = s;
    } else {
        long j = i - 2L * E_RH;
        int kt = (int)(j % TT); long t0 = j / TT;
        int qi = (int)(t0 % QHW); t0 /= QHW;
        int t  = (int)(t0 % TT); int bh = (int)(t0 / TT);
        int tok = t * QHW + qi;
        const float* q = Qf + ((size_t)bh * NTOK + tok) * HD;
        const float* r = rpt + (size_t)(t - kt + 7) * HD;
        float s = 0.f;
        for (int c = 0; c < HD; ++c) s += q[c] * r[c];
        relT[j] = s;
    }
}

// ============================================================
// 4) fused attention: one workgroup = (b, head, 16 query rows)
//    Phase 1: S = scale*Q.K^T + bias  (WMMA, S kept in LDS)
//    Phase 2: rowwise softmax in LDS (unnormalized P, row sums)
//    Phase 3: O = P.V via WMMA (Vt is [c][token]); /l; +q; -> Xattn bf16
// ============================================================
__global__ __launch_bounds__(256) void attn_kernel(
    const bf16_t* __restrict__ Qbf, const bf16_t* __restrict__ Kbf,
    const bf16_t* __restrict__ Vt,  const float* __restrict__ Qf,
    const float* __restrict__ relH, const float* __restrict__ relW,
    const float* __restrict__ relT, bf16_t* __restrict__ Xattn)
{
    extern __shared__ char smem[];
    float* S      = (float*)smem;            // [16][SPAD]
    float* Oacc   = S + 16 * SPAD;           // [16][64]
    float* rowsum = Oacc + 16 * 64;          // [16]
    float* LrelT  = rowsum + 16;             // [16][8]
    float* LrelH  = LrelT + 16 * 8;          // [16][14]
    float* LrelW  = LrelH + 16 * 14;         // [16][14]
    int*   qidxs  = (int*)(LrelW + 16 * 14); // [16]

    int blk = blockIdx.x;
    int qb = blk % 99;                 // 99 query blocks of 16 (last partial)
    int bh = blk / 99;                 // b*12 + head
    int b = bh / NHD, nh = bh % NHD;
    int tid = threadIdx.x;
    int wave = tid >> 5, lane = tid & 31;
    int ln = lane & 15, half = lane >> 4;
    int qbase = qb * 16;

    // --- preload per-query-row rel vectors into LDS ---
    if (tid < 16) {
        int tok = qbase + tid; if (tok >= NTOK) tok = NTOK - 1;
        int qt = tok / QHW, qi = tok % QHW;
        qidxs[tid] = qi;
        const float* rT = relT + (((size_t)bh * TT + qt) * QHW + qi) * TT;
        for (int j = 0; j < TT; ++j) LrelT[tid * 8 + j] = rT[j];
        if (qi > 0) {
            int hw = qi - 1;
            const float* rH = relH + (((size_t)bh * TT + qt) * HW + hw) * 14;
            const float* rW = relW + (((size_t)bh * TT + qt) * HW + hw) * 14;
            for (int j = 0; j < 14; ++j) { LrelH[tid*14+j] = rH[j]; LrelW[tid*14+j] = rW[j]; }
        } else {
            for (int j = 0; j < 14; ++j) { LrelH[tid*14+j] = 0.f; LrelW[tid*14+j] = 0.f; }
        }
    }

    // --- load Q A-fragments once per wave (HD=64 -> 2 k-steps) ---
    AFrag qa[2];
    {
        int tok = qbase + ln; if (tok >= NTOK) tok = NTOK - 1;
        const bf16_t* qrow = Qbf + ((size_t)bh * NTOK + tok) * HD;
        #pragma unroll
        for (int ks = 0; ks < 2; ++ks) {
            int k0 = ks * 32;
            qa[ks].h[0] = *(const v8bf*)(qrow + k0 + half * 8);
            qa[ks].h[1] = *(const v8bf*)(qrow + k0 + 16 + half * 8);
        }
    }
    __syncthreads();

    // --- Phase 1: S tiles (wave-strided over 99 key tiles of 16) ---
    for (int jt = wave; jt < 99; jt += 8) {
        int kbase = jt * 16;
        AFrag kb[2];
        {
            int ktok = kbase + ln; if (ktok >= NTOK) ktok = NTOK - 1;
            const bf16_t* krow = Kbf + ((size_t)bh * NTOK + ktok) * HD;
            #pragma unroll
            for (int ks = 0; ks < 2; ++ks) {
                int k0 = ks * 32;
                kb[ks].h[0] = *(const v8bf*)(krow + k0 + half * 8);
                kb[ks].h[1] = *(const v8bf*)(krow + k0 + 16 + half * 8);
            }
        }
        v8f acc = {};
        acc = wmma_bf16(qa[0], kb[0], acc);
        acc = wmma_bf16(qa[1], kb[1], acc);

        int kk = kbase + ln;                   // this lane's key column
        int kc = (kk < NTOK) ? kk : NTOK - 1;
        int kt = kc / QHW, ki = kc % QHW;
        int kh = 0, kw = 0; bool kvalid = (ki > 0);
        if (kvalid) { kh = (ki - 1) / 14; kw = (ki - 1) % 14; }
        #pragma unroll
        for (int r = 0; r < 8; ++r) {
            int m = half * 8 + r;
            float bias = LrelT[m * 8 + kt];
            if (kvalid && qidxs[m] > 0) bias += LrelH[m*14 + kh] + LrelW[m*14 + kw];
            S[m * SPAD + kk] = acc[r] * SCALE + bias;
        }
    }
    __syncthreads();

    // --- Phase 2: softmax (wave handles 2 rows); zero O accum ---
    for (int i = tid; i < 16 * HD; i += 256) Oacc[i] = 0.f;
    #pragma unroll
    for (int rr = 0; rr < 2; ++rr) {
        int m = wave * 2 + rr;
        float mx = -3.0e38f;
        for (int c = lane; c < NTOK; c += 32) mx = fmaxf(mx, S[m * SPAD + c]);
        for (int off = 16; off > 0; off >>= 1) mx = fmaxf(mx, __shfl_xor(mx, off, 32));
        float sum = 0.f;
        for (int c = lane; c < SPAD; c += 32) {
            float v = (c < NTOK) ? __expf(S[m * SPAD + c] - mx) : 0.f;
            S[m * SPAD + c] = v;
            sum += v;
        }
        for (int off = 16; off > 0; off >>= 1) sum += __shfl_xor(sum, off, 32);
        if (lane == 0) rowsum[m] = sum;
    }
    __syncthreads();

    // --- Phase 3: O = P.V  (wave-strided over 50 key tiles of 32) ---
    v8f oacc[4] = {};
    for (int j = wave; j < 50; j += 8) {
        int kb0 = j * 32;
        AFrag pa;                              // P A-fragment from LDS (f32->bf16)
        {
            const float* srow = S + (size_t)ln * SPAD + kb0;
            float tmp[16];
            #pragma unroll
            for (int i2 = 0; i2 < 8; ++i2) tmp[i2]     = srow[half * 8 + i2];
            #pragma unroll
            for (int i2 = 0; i2 < 8; ++i2) tmp[8 + i2] = srow[16 + half * 8 + i2];
            #pragma unroll
            for (int i2 = 0; i2 < 16; ++i2) pa.v[i2] = (bf16_t)tmp[i2];
        }
        #pragma unroll
        for (int g = 0; g < 4; ++g) {          // 4 column groups cover HD=64
            AFrag vb;
            const bf16_t* vrow = Vt + ((size_t)bh * HD + g * 16 + ln) * SPAD + kb0;
            vb.h[0] = *(const v8bf*)(vrow + half * 8);
            vb.h[1] = *(const v8bf*)(vrow + 16 + half * 8);
            oacc[g] = wmma_bf16(pa, vb, oacc[g]);
        }
    }
    #pragma unroll
    for (int g = 0; g < 4; ++g)
        #pragma unroll
        for (int r = 0; r < 8; ++r)
            atomicAdd(&Oacc[(half * 8 + r) * HD + g * 16 + ln], oacc[g][r]);
    __syncthreads();

    // --- epilogue: normalize, +q residual, write bf16 [b, tok, nh*64+c] ---
    for (int i = tid; i < 16 * HD; i += 256) {
        int m = i >> 6, c = i & 63;
        int tok = qbase + m;
        if (tok < NTOK) {
            float val = Oacc[i] / rowsum[m] + Qf[((size_t)bh * NTOK + tok) * HD + c];
            Xattn[((size_t)(b * NTOK + tok)) * DIM + nh * HD + c] = (bf16_t)val;
        }
    }
}

// ============================================================
// 5) output projection GEMM: out[R,O] = X[R,:].Wp[O,:] + b_proj[O]  (f32 out)
// ============================================================
__global__ __launch_bounds__(256) void proj_gemm_kernel(
    const bf16_t* __restrict__ X, const bf16_t* __restrict__ W,
    const float* __restrict__ bp, float* __restrict__ out)
{
    const int MT = MROWS / 16;      // 197
    const int NG = DIM / 64;        // 12
    int gwave = blockIdx.x * 8 + (threadIdx.x >> 5);
    if (gwave >= MT * NG) return;
    int mt = gwave % MT, ng = gwave / MT;
    int lane = threadIdx.x & 31, ln = lane & 15, half = lane >> 4;
    int n0 = ng * 64;
    const bf16_t* arow = X + (size_t)(mt * 16 + ln) * DIM;

    v8f acc[4] = {};
    for (int k0 = 0; k0 < DIM; k0 += 32) {
        AFrag a;
        a.h[0] = *(const v8bf*)(arow + k0 + half * 8);
        a.h[1] = *(const v8bf*)(arow + k0 + 16 + half * 8);
        #pragma unroll
        for (int g = 0; g < 4; ++g) {
            AFrag bfr;
            const bf16_t* brow = W + (size_t)(n0 + g * 16 + ln) * DIM + k0;
            bfr.h[0] = *(const v8bf*)(brow + half * 8);
            bfr.h[1] = *(const v8bf*)(brow + 16 + half * 8);
            acc[g] = wmma_bf16(a, bfr, acc[g]);
        }
    }
    #pragma unroll
    for (int g = 0; g < 4; ++g) {
        int O = n0 + g * 16 + ln;
        float bias = bp[O];
        #pragma unroll
        for (int r = 0; r < 8; ++r) {
            int R = mt * 16 + half * 8 + r;
            out[(size_t)R * DIM + O] = acc[g][r] + bias;
        }
    }
}

// ============================================================
// launcher
// ============================================================
extern "C" void kernel_launch(void* const* d_in, const int* in_sizes, int n_in,
                              void* d_out, int out_size, void* d_ws, size_t ws_size,
                              hipStream_t stream) {
    const float* query     = (const float*)d_in[0];
    const float* key_value = (const float*)d_in[1];
    const float* w_qkv     = (const float*)d_in[2];
    const float* b_qkv     = (const float*)d_in[3];
    const float* w_proj    = (const float*)d_in[4];
    const float* b_proj    = (const float*)d_in[5];
    const float* rel_pos_h = (const float*)d_in[6];
    const float* rel_pos_w = (const float*)d_in[7];
    const float* rel_pos_t = (const float*)d_in[8];
    float* out = (float*)d_out;

    // workspace carve-up (256B aligned)
    char* ws = (char*)d_ws;
    size_t off = 0;
    auto carve = [&](size_t bytes) -> char* {
        char* p = ws + off;
        off = (off + bytes + 255) & ~(size_t)255;
        return p;
    };
    bf16_t* Xq    = (bf16_t*)carve((size_t)E_X  * 2);
    bf16_t* Xkv   = (bf16_t*)carve((size_t)E_X  * 2);
    bf16_t* Wq    = (bf16_t*)carve((size_t)E_WQ * 2);
    bf16_t* Wp    = (bf16_t*)carve((size_t)E_WP * 2);
    bf16_t* Qbf   = (bf16_t*)carve((size_t)E_X  * 2);
    bf16_t* Kbf   = (bf16_t*)carve((size_t)E_X  * 2);
    bf16_t* Vt    = (bf16_t*)carve((size_t)E_VT * 2);
    float*  Qf    = (float*) carve((size_t)E_X  * 4);
    float*  relH  = (float*) carve((size_t)E_RH * 4);
    float*  relW  = (float*) carve((size_t)E_RH * 4);
    float*  relT  = (float*) carve((size_t)E_RT * 4);
    bf16_t* Xattn = (bf16_t*)carve((size_t)E_X  * 2);
    (void)ws_size; (void)in_sizes; (void)n_in; (void)out_size;

    // 1) convert + zero Vt
    {
        long total = (long)E_X + E_X + E_WQ + E_WP + E_VT;
        int blocks = (int)((total + 255) / 256);
        convert_kernel<<<blocks, 256, 0, stream>>>(query, key_value, w_qkv, w_proj,
                                                   Xq, Xkv, Wq, Wp, Vt);
    }
    // 2) QKV GEMM
    {
        int waves = (MROWS / 16) * (DIM3 / 64);           // 7092
        int blocks = (waves + 7) / 8;
        qkv_gemm_kernel<<<blocks, 256, 0, stream>>>(Xq, Xkv, Wq, b_qkv,
                                                    Qbf, Qf, Kbf, Vt);
    }
    // 3) rel-pos bias precompute
    {
        long total = 2L * E_RH + E_RT;
        int blocks = (int)((total + 255) / 256);
        relbias_kernel<<<blocks, 256, 0, stream>>>(Qf, rel_pos_h, rel_pos_w, rel_pos_t,
                                                   relH, relW, relT);
    }
    // 4) fused attention
    {
        size_t smem = (size_t)(16 * SPAD + 16 * HD + 16 + 16*8 + 16*14 + 16*14) * 4
                    + 16 * sizeof(int);                   // ~108.9 KB (LDS=320KB/WGP)
        int blocks = 2 * NHD * 99;                        // 2376
        attn_kernel<<<blocks, 256, smem, stream>>>(Qbf, Kbf, Vt, Qf,
                                                   relH, relW, relT, Xattn);
    }
    // 5) output projection
    {
        int waves = (MROWS / 16) * (DIM / 64);            // 2364
        int blocks = (waves + 7) / 8;
        proj_gemm_kernel<<<blocks, 256, 0, stream>>>(Xattn, Wp, b_proj, out);
    }
}
